// ResidualBottleneck_89816356094475
// MI455X (gfx1250) — matrix operand order repaired and checked by reading, not compiled, not run on hardware
//
#include <hip/hip_runtime.h>
#include <cstdint>
#include <cstddef>

typedef __attribute__((ext_vector_type(16))) __bf16 v16bf;
typedef __attribute__((ext_vector_type(8)))  __bf16 v8bf;
typedef __attribute__((ext_vector_type(4)))  __bf16 v4bf;
typedef __attribute__((ext_vector_type(8)))  float  v8f;

// Problem constants
#define BB   16
#define DD   512
#define TT   4096
#define KK   1024
#define NQ   8
#define NROW (BB * TT)            // 65536
#define BDT  ((size_t)BB * DD * TT)

// LDS layout for the layer kernel
#define AS 528                     // bf16 stride for 64x512 A tile (bank-conflict pad)
#define BS 528                     // bf16 stride for 16x512 B tile
#define QS 516                     // f32 stride for 32x512 q staging (reuses A region)
#define A_BYTES   (64 * AS * 2)    // 67584
#define B_BYTES   (16 * BS * 2)    // 16896
#define SMEM_A    0
#define SMEM_B0   (SMEM_A + A_BYTES)
#define SMEM_B1   (SMEM_B0 + B_BYTES)
#define SMEM_NRM  (SMEM_B1 + B_BYTES)
#define SMEM_IDX  (SMEM_NRM + 1024 * 4)
#define SMEM_RED  (SMEM_IDX + 64 * 4)
#define SMEM_TOT  (SMEM_RED + 128 * 4)   // ~106 KB, fits 320 KB/WGP

// ---------------------------------------------------------------------------
// Prep 1: x [B, D, T] -> resid [B*T, D]  (tiled transpose, coalesced both sides)
// ---------------------------------------------------------------------------
__global__ __launch_bounds__(256) void xpose_in_kernel(const float* __restrict__ x,
                                                       float* __restrict__ resid) {
  __shared__ float tile[32][33];
  const int b  = blockIdx.z;
  const int t0 = blockIdx.x * 32;
  const int d0 = blockIdx.y * 32;
  const int tx = threadIdx.x, ty = threadIdx.y;   // 32 x 8
#pragma unroll
  for (int j = 0; j < 4; ++j) {
    const int d = d0 + ty + j * 8;
    tile[ty + j * 8][tx] = x[((size_t)b * DD + d) * TT + t0 + tx];
  }
  __syncthreads();
#pragma unroll
  for (int j = 0; j < 4; ++j) {
    const int t = t0 + ty + j * 8;
    resid[((size_t)b * TT + t) * DD + d0 + tx] = tile[tx][ty + j * 8];
  }
}

// ---------------------------------------------------------------------------
// Prep 2: codebooks fp32 -> bf16 copy + per-code squared norms
// ---------------------------------------------------------------------------
__global__ __launch_bounds__(128) void cb_prep_kernel(const float* __restrict__ cb,
                                                      __bf16* __restrict__ cbbf,
                                                      float* __restrict__ norms) {
  __shared__ float red[128];
  const int cg  = blockIdx.x;          // 0 .. NQ*KK-1
  const int tid = threadIdx.x;
  const float4 v = *(const float4*)(cb + (size_t)cg * DD + tid * 4);
  v4bf p;
  p[0] = (__bf16)v.x; p[1] = (__bf16)v.y; p[2] = (__bf16)v.z; p[3] = (__bf16)v.w;
  *(v4bf*)(cbbf + (size_t)cg * DD + tid * 4) = p;
  red[tid] = v.x * v.x + v.y * v.y + v.z * v.z + v.w * v.w;
  __syncthreads();
  for (int s = 64; s > 0; s >>= 1) {
    if (tid < s) red[tid] += red[tid + s];
    __syncthreads();
  }
  if (tid == 0) norms[cg] = red[0];
}

// ---------------------------------------------------------------------------
// B-fragment fetch from LDS for a 32x16 bf16 WMMA operand, contiguous k order (as A):
// lanes 0-15 hold K 0..15 of column n=lane, lanes 16-31 hold K 16..31.
// ---------------------------------------------------------------------------
__device__ __forceinline__ v16bf load_bfrag(const __bf16* bb, int n, int half, int kc) {
  const int base = n * BS + kc * 32 + half * 16;
  v8bf lo = *(const v8bf*)(bb + base);
  v8bf hi = *(const v8bf*)(bb + base + 8);
  return __builtin_shufflevector(lo, hi, 0,1,2,3,4,5,6,7,8,9,10,11,12,13,14,15);
}

// ---------------------------------------------------------------------------
// Per-layer RVQ kernel: bf16 WMMA score GEMM + argmin + fp32 residual update
// ---------------------------------------------------------------------------
__global__ __launch_bounds__(128) void rvq_layer_kernel(
    const float*  __restrict__ cbf32,   // this layer's codebook, fp32 [K, D]
    const __bf16* __restrict__ cbbf,    // bf16 copy [K, D]
    const float*  __restrict__ norms,   // |e_k|^2, [K]
    float* __restrict__ resid,          // [NROW, D], updated in place
    float* __restrict__ out,            // d_out "out" region, [B, D, T]
    float*            qout,             // q1/q2 region or nullptr
    float* __restrict__ loss_accum,     // single f32 accumulator
    int first) {
  __shared__ alignas(16) unsigned char smem[SMEM_TOT];
  __bf16* Abuf = (__bf16*)(smem + SMEM_A);
  __bf16* Bbuf0 = (__bf16*)(smem + SMEM_B0);
  __bf16* Bbuf1 = (__bf16*)(smem + SMEM_B1);
  float*  nrm  = (float*)(smem + SMEM_NRM);
  int*    idxs = (int*)(smem + SMEM_IDX);
  float*  red  = (float*)(smem + SMEM_RED);

  const int tid  = threadIdx.x;
  const int wave = tid >> 5;
  const int lane = tid & 31;
  const int half = lane >> 4;
  const int m    = lane & 15;          // A row (within 16) / B column (within 16)
  const int row0 = blockIdx.x * 64;

  // ---- stage residual tile (fp32 -> bf16, float4 loads, b64 LDS stores) ----
  for (int i = tid; i < 64 * 128; i += 128) {
    const int r = i >> 7, d4 = (i & 127) * 4;
    const float4 v = *(const float4*)(resid + (size_t)(row0 + r) * DD + d4);
    v4bf p;
    p[0] = (__bf16)v.x; p[1] = (__bf16)v.y; p[2] = (__bf16)v.z; p[3] = (__bf16)v.w;
    *(v4bf*)(Abuf + r * AS + d4) = p;
  }
  for (int i = tid; i < KK; i += 128) nrm[i] = norms[i];
  __syncthreads();

  // ---- build this wave's 16x512 A operand in registers (16 x v16bf) ----
  v16bf a[16];
#pragma unroll
  for (int kc = 0; kc < 16; ++kc) {
    // A 16x32 in this code's contiguous k order (B matches): lane half 0 holds K 0..15, half 1 K 16..31
    const int base = (wave * 16 + m) * AS + kc * 32 + half * 16;
    v8bf lo = *(const v8bf*)(Abuf + base);
    v8bf hi = *(const v8bf*)(Abuf + base + 8);
    a[kc] = __builtin_shufflevector(lo, hi, 0,1,2,3,4,5,6,7,8,9,10,11,12,13,14,15);
  }

  float minv[8];
  int   mini[8];
#pragma unroll
  for (int j = 0; j < 8; ++j) { minv[j] = 3.4e38f; mini[j] = 0; }

  // ---- async-to-LDS issue of one 16-code bf16 chunk (8 x b128 per thread) ----
  const unsigned lds_b0 = (unsigned)(uintptr_t)Bbuf0;
  const unsigned lds_b1 = (unsigned)(uintptr_t)Bbuf1;
  const int crow = tid >> 3;                 // code row 0..15
  const int seg  = (tid & 7) * 128;          // byte offset within 1KB row
#define ISSUE_CHUNK(ldsbase, nc)                                                        \
  {                                                                                     \
    const char* g = (const char*)(cbbf + (size_t)((nc) * 16 + crow) * DD) + seg;        \
    unsigned ld = (ldsbase) + (unsigned)(crow * (BS * 2)) + (unsigned)seg;              \
    _Pragma("unroll")                                                                   \
    for (int ii = 0; ii < 8; ++ii) {                                                    \
      asm volatile("global_load_async_to_lds_b128 %0, %1, off"                          \
                   :: "v"(ld + ii * 16), "v"(g + ii * 16) : "memory");                  \
    }                                                                                   \
  }

  ISSUE_CHUNK(lds_b0, 0);
  for (int nc = 0; nc < 64; ++nc) {
    const __bf16* bb = (nc & 1) ? Bbuf1 : Bbuf0;
    if (nc + 1 < 64) {
      if (nc & 1) { ISSUE_CHUNK(lds_b0, nc + 1); } else { ISSUE_CHUNK(lds_b1, nc + 1); }
      asm volatile("s_wait_asynccnt 0x8" ::: "memory");
    } else {
      asm volatile("s_wait_asynccnt 0x0" ::: "memory");
    }
    __syncthreads();

    const int cbase = nc * 16;
    const float nn = nrm[cbase + m];         // this lane's column norm
    v8f c = {};
    // Group-of-2 interleave: both B fragments are live before the first WMMA,
    // forcing two register sets so the second pair of ds_loads overlaps the
    // first WMMA (partial dscnt wait instead of 0).
#pragma unroll
    for (int kc = 0; kc < 16; kc += 2) {
      v16bf bx = load_bfrag(bb, m, half, kc);
      v16bf by = load_bfrag(bb, m, half, kc + 1);
      c = __builtin_amdgcn_wmma_f32_16x16x32_bf16(false, a[kc], false, bx,
                                                  (short)0, c, false, false);
      c = __builtin_amdgcn_wmma_f32_16x16x32_bf16(false, a[kc + 1], false, by,
                                                  (short)0, c, false, false);
    }
#pragma unroll
    for (int j = 0; j < 8; ++j) {
      const float score = nn - 2.0f * c[j];  // |e|^2 - 2 r.e  (|r|^2 dropped)
      if (score < minv[j]) { minv[j] = score; mini[j] = cbase + m; }
    }
    __syncthreads();                          // protect buffer being refilled next iter
  }

  // ---- cross-lane argmin over the 16 N-lanes (first-min tie-break) ----
#pragma unroll
  for (int j = 0; j < 8; ++j) {
#pragma unroll
    for (int off = 8; off >= 1; off >>= 1) {
      const float ov = __shfl_xor(minv[j], off, 32);
      const int   oi = __shfl_xor(mini[j], off, 32);
      if (ov < minv[j] || (ov == minv[j] && oi < mini[j])) { minv[j] = ov; mini[j] = oi; }
    }
  }
  if (m == 0) {
#pragma unroll
    for (int j = 0; j < 8; ++j) idxs[wave * 16 + half * 8 + j] = mini[j];
  }
  __syncthreads();

  // ---- gather + residual update + transposed output writes ----
  float* qstage = (float*)(smem + SMEM_A);    // A tile region reused (regs hold A)
  const int bidx  = row0 / TT;
  const int tbas0 = row0 % TT;
  float lsum = 0.0f;

  for (int h = 0; h < 2; ++h) {
    for (int r8 = 0; r8 < 32; ++r8) {
      const int r = h * 32 + r8;
      const int code = idxs[r];
      const float4 q4 = *(const float4*)(cbf32 + (size_t)code * DD + tid * 4);
      float* rp = resid + (size_t)(row0 + r) * DD + tid * 4;
      float4 r4 = *(const float4*)rp;
      float4 nv; nv.x = r4.x - q4.x; nv.y = r4.y - q4.y; nv.z = r4.z - q4.z; nv.w = r4.w - q4.w;
      *(float4*)rp = nv;
      lsum += nv.x * nv.x + nv.y * nv.y + nv.z * nv.z + nv.w * nv.w;
      *(float4*)(qstage + r8 * QS + tid * 4) = q4;
    }
    __syncthreads();
    const int tbase = tbas0 + h * 32;
    for (int i = 0; i < 128; ++i) {           // 512 d x 32 t, 32-float coalesced runs
      const int lin = i * 128 + tid;
      const int d = lin >> 5, ttl = lin & 31;
      const float v = qstage[ttl * QS + d];
      const size_t o = (size_t)bidx * DD * TT + (size_t)d * TT + tbase + ttl;
      if (first) { out[o] = v; qout[o] = v; }
      else       { out[o] += v; if (qout) qout[o] = v; }
    }
    __syncthreads();
  }

  // ---- block loss reduce + global atomic accumulate ----
  red[tid] = lsum;
  __syncthreads();
  for (int s = 64; s > 0; s >>= 1) {
    if (tid < s) red[tid] += red[tid + s];
    __syncthreads();
  }
  if (tid == 0) atomicAdd(loss_accum, red[0]);
}

__global__ void finalize_kernel(const float* __restrict__ loss, float* __restrict__ com) {
  // com = sum over 8 layers of per-layer sums / (8 * NROW * D), weight = 1.0
  com[0] = loss[0] * (1.0f / 268435456.0f);
}

// ---------------------------------------------------------------------------
extern "C" void kernel_launch(void* const* d_in, const int* in_sizes, int n_in,
                              void* d_out, int out_size, void* d_ws, size_t ws_size,
                              hipStream_t stream) {
  (void)in_sizes; (void)n_in; (void)out_size; (void)ws_size;
  const float* x    = (const float*)d_in[0];   // [B, D, T]
  const float* cbks = (const float*)d_in[1];   // [NQ, K, D]
  float* outp = (float*)d_out;

  // Workspace carve (~136 MB): resid fp32, bf16 codebooks, norms, loss accum
  char* ws = (char*)d_ws;
  float*  resid  = (float*)ws;                                   // NROW*D*4 = 128 MB
  __bf16* cbbf   = (__bf16*)(ws + (size_t)NROW * DD * 4);        // NQ*K*D*2 = 8 MB
  float*  norms  = (float*)((char*)cbbf + (size_t)NQ * KK * DD * 2);
  float*  lossac = (float*)((char*)norms + (size_t)NQ * KK * 4);

  hipMemsetAsync(lossac, 0, 64, stream);

  dim3 tg(TT / 32, DD / 32, BB);
  xpose_in_kernel<<<tg, dim3(32, 8), 0, stream>>>(x, resid);
  cb_prep_kernel<<<NQ * KK, 128, 0, stream>>>(cbks, cbbf, norms);

  for (int layer = 0; layer < NQ; ++layer) {
    float* qout = nullptr;
    if (layer == 0) qout = outp + BDT;        // q1
    if (layer == 1) qout = outp + 2 * BDT;    // q2
    rvq_layer_kernel<<<NROW / 64, 128, 0, stream>>>(
        cbks + (size_t)layer * KK * DD,
        cbbf + (size_t)layer * KK * DD,
        norms + (size_t)layer * KK,
        resid, outp, qout, lossac, layer == 0 ? 1 : 0);
  }
  finalize_kernel<<<1, 1, 0, stream>>>(lossac, outp + 3 * BDT);
}
